// EngramMemory_62002147885489
// MI455X (gfx1250) — matrix operand order repaired and checked by reading, not compiled, not run on hardware
//
#include <hip/hip_runtime.h>
#include <cstddef>
#include <cstdint>

// ---------------------------------------------------------------------------
// EngramMemory forward for MI455X (gfx1250, wave32, WMMA).
// GEMMs: error-compensated bf16x3 (hi/lo split) on v_wmma_f32_16x16x32_bf16,
// with double-buffered GLOBAL_LOAD_ASYNC_TO_LDS_B128 tile staging.
// ---------------------------------------------------------------------------

typedef __attribute__((ext_vector_type(16))) __bf16 v16bf;
typedef __attribute__((ext_vector_type(8)))  float  v8f;

#define D_EMB 512
#define HID_DIM 1536
#define CONCAT_DIM 4096
#define TABLE_SZ 50000
#define SEQ_LEN 2048

// ---------------- CDNA5 async copy helpers (gfx1250) ----------------
// Per-lane 16B copy global -> LDS, tracked with ASYNCcnt (no VGPR round trip).
__device__ __forceinline__ void async_copy16(unsigned lds_addr, const void* gsrc) {
    asm volatile("global_load_async_to_lds_b128 %0, %1, off"
                 :: "v"(lds_addr), "v"(gsrc)
                 : "memory");
}
__device__ __forceinline__ unsigned lds_off(const void* p) {
    // generic LDS pointer: addr[31:0] is the LDS byte address (aperture rules)
    return (unsigned)(size_t)p;
}
__device__ __forceinline__ void wait_async0() {
    asm volatile("s_wait_asynccnt 0x0" ::: "memory");
}
__device__ __forceinline__ void wait_async8() {
    asm volatile("s_wait_asynccnt 0x8" ::: "memory");
}

// ---------------- bf16 helpers ----------------
__device__ __forceinline__ unsigned short f32_to_bf16(float x) {
    unsigned u = __float_as_uint(x);
    unsigned r = u + 0x7fffu + ((u >> 16) & 1u);   // round-to-nearest-even
    return (unsigned short)(r >> 16);
}
__device__ __forceinline__ float bf16_to_f32(unsigned short h) {
    return __uint_as_float(((unsigned)h) << 16);
}
__device__ __forceinline__ void split_bf16(float x, unsigned short& hi, unsigned short& lo) {
    hi = f32_to_bf16(x);
    lo = f32_to_bf16(x - bf16_to_f32(hi));
}

// ---------------- block reduction (256 threads) ----------------
__device__ __forceinline__ float block_reduce_sum256(float x, float* sh) {
    int t = threadIdx.x;
    sh[t] = x; __syncthreads();
    for (int o = 128; o > 0; o >>= 1) {
        if (t < o) sh[t] += sh[t + o];
        __syncthreads();
    }
    float r = sh[0]; __syncthreads();
    return r;
}

// ---------------- 1) split weight into bf16 hi/lo, TRANSPOSED [N,K] ----------
// Pre-transposing B lets GEMM tile staging be plain strided copies (async-able)
// while keeping the per-lane column layout the WMMA B operand wants.
__global__ void __launch_bounds__(256)
split_wT_kernel(const float* __restrict__ w, unsigned short* __restrict__ hiT,
                unsigned short* __restrict__ loT, int K, int N) {
    int i = blockIdx.x * 256 + threadIdx.x;
    if (i < K * N) {
        int k = i / N, n = i - k * N;
        unsigned short h, l;
        split_bf16(w[i], h, l);
        hiT[(size_t)n * K + k] = h;
        loT[(size_t)n * K + k] = l;
    }
}

// ---------------- 2) hash + gather (async) + LayerNorm(CONCAT) + split -------
struct Mults { unsigned m[8]; };

__global__ void __launch_bounds__(256)
retrieve_ln_split_kernel(const int* __restrict__ tok, const float* __restrict__ tables,
                         const float* __restrict__ gam, const float* __restrict__ bet,
                         unsigned short* __restrict__ ahi, unsigned short* __restrict__ alo,
                         Mults mu, int S) {
    __shared__ float row[CONCAT_DIM];
    __shared__ float red[256];
    const int idx = blockIdx.x;          // b*S + s
    const int b = idx / S, s = idx - b * S;
    const int t = threadIdx.x;
    const int p = t >> 5;                // each wave owns one (order, head) part
    const int lane = t & 31;
    const int n = 2 + (p >> 2);          // p<4 -> bigram, p>=4 -> trigram

    if (s >= n - 1) {
        unsigned h = 0;
        for (int j = 0; j < n; ++j) {
            unsigned tk = (unsigned)tok[b * S + s - (n - 1) + j];
            h = (h * mu.m[p] + tk) % (unsigned)TABLE_SZ;
        }
        const float* src = tables + ((size_t)p * TABLE_SZ + h) * D_EMB;
        #pragma unroll
        for (int j = 0; j < 4; ++j) {
            const int c = (lane + 32 * j) * 4;          // float index in row
            async_copy16(lds_off(&row[p * D_EMB + c]), src + c);
        }
        wait_async0();
    } else {
        #pragma unroll
        for (int j = 0; j < 4; ++j) {
            const int c = (lane + 32 * j) * 4;
            *(float4*)&row[p * D_EMB + c] = make_float4(0.f, 0.f, 0.f, 0.f);
        }
    }
    __syncthreads();

    float s1 = 0.f, s2 = 0.f;
    for (int i = t; i < CONCAT_DIM; i += 256) { float x = row[i]; s1 += x; s2 += x * x; }
    float tot  = block_reduce_sum256(s1, red);
    float tot2 = block_reduce_sum256(s2, red);
    float mean = tot * (1.0f / CONCAT_DIM);
    float inv  = rsqrtf(tot2 * (1.0f / CONCAT_DIM) - mean * mean + 1e-5f);

    const size_t base = (size_t)idx * CONCAT_DIM;
    for (int i = t; i < CONCAT_DIM; i += 256) {
        float y = (row[i] - mean) * inv * gam[i] + bet[i];
        unsigned short h, l; split_bf16(y, h, l);
        ahi[base + i] = h; alo[base + i] = l;
    }
}

// ---------------- 3) plain LayerNorm + bf16 split (hidden, W=1536) -----------
template <int W>
__global__ void __launch_bounds__(256)
ln_split_kernel(const float* __restrict__ x, const float* __restrict__ gam,
                const float* __restrict__ bet, unsigned short* __restrict__ hi,
                unsigned short* __restrict__ lo) {
    __shared__ float row[W];
    __shared__ float red[256];
    const int r = blockIdx.x, t = threadIdx.x;
    const float* xr = x + (size_t)r * W;
    for (int i = t; i < W; i += 256) row[i] = xr[i];
    __syncthreads();
    float s1 = 0.f, s2 = 0.f;
    for (int i = t; i < W; i += 256) { float v = row[i]; s1 += v; s2 += v * v; }
    float tot  = block_reduce_sum256(s1, red);
    float tot2 = block_reduce_sum256(s2, red);
    float mean = tot * (1.0f / W);
    float inv  = rsqrtf(tot2 * (1.0f / W) - mean * mean + 1e-5f);
    const size_t base = (size_t)r * W;
    for (int i = t; i < W; i += 256) {
        float y = (row[i] - mean) * inv * gam[i] + bet[i];
        unsigned short h, l; split_bf16(y, h, l);
        hi[base + i] = h; lo[base + i] = l;
    }
}

// ---------------- 4) bf16x3 WMMA GEMM: C[M,N] = A[M,K] @ BT[N,K]^T -----------
__device__ __forceinline__ v16bf load_frag(const unsigned short* p0,
                                           const unsigned short* p1) {
    union { uint4 q[2]; v16bf v; } u;
    u.q[0] = *(const uint4*)p0;
    u.q[1] = *(const uint4*)p1;
    return u.v;
}

__global__ void __launch_bounds__(256)
gemm_bf16x3_kernel(const unsigned short* __restrict__ Ahi, const unsigned short* __restrict__ Alo,
                   const unsigned short* __restrict__ BThi, const unsigned short* __restrict__ BTlo,
                   float* __restrict__ C, int M, int N, int K) {
    // LDS rows padded to 40 u16 (80 B): rows stay 16B-aligned, banks spread.
    // Double-buffered for async pipelining: 4 arrays x 2 x 10 KB = 80 KB of 320 KB.
    __shared__ alignas(16) unsigned short sAh[2][128][40];
    __shared__ alignas(16) unsigned short sAl[2][128][40];
    __shared__ alignas(16) unsigned short sBh[2][128][40];   // [n][k]
    __shared__ alignas(16) unsigned short sBl[2][128][40];

    const int t    = threadIdx.x;
    const int lane = t & 31, wave = t >> 5;
    const int wm = wave & 3, wn = wave >> 2;              // 4 x 2 wave grid
    const int half = lane >> 4, l15 = lane & 15;
    const int mBlock = blockIdx.x * 128, nBlock = blockIdx.y * 128;

    v8f acc[2][4];
    #pragma unroll
    for (int mi = 0; mi < 2; ++mi)
        #pragma unroll
        for (int ni = 0; ni < 4; ++ni)
            #pragma unroll
            for (int r = 0; r < 8; ++r) acc[mi][ni][r] = 0.0f;

    // staging map: thread -> (row 0..127, 16-element half 0/16)
    const int row  = t >> 1;
    const int colh = (t & 1) * 16;
    const size_t aBase = (size_t)(mBlock + row) * K + colh;
    const size_t bBase = (size_t)(nBlock + row) * K + colh;

    auto issue = [&](int buf, int k0) {
        async_copy16(lds_off(&sAh[buf][row][colh]),     Ahi  + aBase + k0);
        async_copy16(lds_off(&sAh[buf][row][colh + 8]), Ahi  + aBase + k0 + 8);
        async_copy16(lds_off(&sAl[buf][row][colh]),     Alo  + aBase + k0);
        async_copy16(lds_off(&sAl[buf][row][colh + 8]), Alo  + aBase + k0 + 8);
        async_copy16(lds_off(&sBh[buf][row][colh]),     BThi + bBase + k0);
        async_copy16(lds_off(&sBh[buf][row][colh + 8]), BThi + bBase + k0 + 8);
        async_copy16(lds_off(&sBl[buf][row][colh]),     BTlo + bBase + k0);
        async_copy16(lds_off(&sBl[buf][row][colh + 8]), BTlo + bBase + k0 + 8);
    };

    const int T = K / 32;
    issue(0, 0);
    int buf = 0;
    for (int i = 0; i < T; ++i) {
        if (i + 1 < T) {
            issue(buf ^ 1, (i + 1) * 32);   // prefetch next tile into other buffer
            wait_async8();                  // current tile's 8 copies have landed
        } else {
            wait_async0();
        }
        __syncthreads();

        // ---- fragments (ISA 7.12.2 layouts) ----
        v16bf afh[2], afl[2], bfh[4], bfl[4];
        #pragma unroll
        for (int mi = 0; mi < 2; ++mi) {
            const int m = wm * 32 + mi * 16 + l15;
            afh[mi] = load_frag(&sAh[buf][m][half * 8], &sAh[buf][m][16 + half * 8]);
            afl[mi] = load_frag(&sAl[buf][m][half * 8], &sAl[buf][m][16 + half * 8]);
        }
        #pragma unroll
        for (int ni = 0; ni < 4; ++ni) {
            const int n = wn * 64 + ni * 16 + l15;
            bfh[ni] = load_frag(&sBh[buf][n][half * 8], &sBh[buf][n][16 + half * 8]);
            bfl[ni] = load_frag(&sBl[buf][n][half * 8], &sBl[buf][n][16 + half * 8]);
        }
        // ---- bf16x3: Ah*Bh + Ah*Bl + Al*Bh, f32 accumulate ----
        #pragma unroll
        for (int mi = 0; mi < 2; ++mi)
            #pragma unroll
            for (int ni = 0; ni < 4; ++ni) {
                acc[mi][ni] = __builtin_amdgcn_wmma_f32_16x16x32_bf16(
                    false, afh[mi], false, bfh[ni], (short)0, acc[mi][ni], false, false);
                acc[mi][ni] = __builtin_amdgcn_wmma_f32_16x16x32_bf16(
                    false, afh[mi], false, bfl[ni], (short)0, acc[mi][ni], false, false);
                acc[mi][ni] = __builtin_amdgcn_wmma_f32_16x16x32_bf16(
                    false, afl[mi], false, bfh[ni], (short)0, acc[mi][ni], false, false);
            }
        if (i + 1 < T) __syncthreads();     // protect buf before it is re-filled
        buf ^= 1;
    }

    // ---- epilogue: C/D layout — lane<16: M=r, lane>=16: M=8+r; N=lane&15 ----
    #pragma unroll
    for (int mi = 0; mi < 2; ++mi) {
        const int mBase = mBlock + wm * 32 + mi * 16 + half * 8;
        #pragma unroll
        for (int ni = 0; ni < 4; ++ni) {
            const int gn = nBlock + wn * 64 + ni * 16 + l15;
            #pragma unroll
            for (int r = 0; r < 8; ++r)
                C[(size_t)(mBase + r) * N + gn] = acc[mi][ni][r];
        }
    }
}

// ---------------- 5) cosine gate + mem = gate * v ----------------
__global__ void __launch_bounds__(256)
gate_mem_kernel(const float* __restrict__ q, const float* __restrict__ k,
                const float* __restrict__ v, float* __restrict__ mem,
                float* __restrict__ gate) {
    __shared__ float red[256];
    const int r = blockIdx.x, t = threadIdx.x;
    const float* qr = q + (size_t)r * D_EMB;
    const float* kr = k + (size_t)r * D_EMB;
    const float* vr = v + (size_t)r * D_EMB;
    float dq = 0.f, dk = 0.f, dd = 0.f;
    for (int i = t; i < D_EMB; i += 256) {
        float a = qr[i], c = kr[i];
        dq += a * a; dk += c * c; dd += a * c;
    }
    dq = block_reduce_sum256(dq, red);
    dk = block_reduce_sum256(dk, red);
    dd = block_reduce_sum256(dd, red);
    float nq = fmaxf(sqrtf(dq), 1e-12f);
    float nk = fmaxf(sqrtf(dk), 1e-12f);
    float gl = dd / (nq * nk) * 22.62741699796952f;   // sqrt(512)
    float gt = 1.0f / (1.0f + expf(-gl));
    for (int i = t; i < D_EMB; i += 256) mem[(size_t)r * D_EMB + i] = gt * vr[i];
    if (t == 0) gate[r] = gt;
}

// ---------------- 6) causal depthwise conv (K=4) + residual + bf16 split -----
__global__ void __launch_bounds__(256)
conv_split_kernel(const float* __restrict__ mem, const float* __restrict__ cw,
                  const float* __restrict__ cb, unsigned short* __restrict__ yhi,
                  unsigned short* __restrict__ ylo, int S) {
    const long long i = (long long)blockIdx.x * 256 + threadIdx.x;
    const int d = (int)(i % D_EMB);
    const long long sd = i / D_EMB;
    const int s = (int)(sd % S);
    float acc = mem[i] + cb[d];                     // residual + bias
    #pragma unroll
    for (int j = 0; j < 4; ++j) {
        int ts = s + j - 3;                         // conv_w[3] multiplies current
        if (ts >= 0) acc += cw[j * D_EMB + d] * mem[i + (long long)(j - 3) * D_EMB];
    }
    unsigned short h, l; split_bf16(acc, h, l);
    yhi[i] = h; ylo[i] = l;
}

// ---------------- 7) out = LayerNorm(tmp + bo) ----------------
__global__ void __launch_bounds__(256)
bias_ln_kernel(const float* __restrict__ tmp, const float* __restrict__ bo,
               const float* __restrict__ gam, const float* __restrict__ bet,
               float* __restrict__ out) {
    __shared__ float row[D_EMB];
    __shared__ float red[256];
    const int r = blockIdx.x, t = threadIdx.x;
    for (int i = t; i < D_EMB; i += 256) row[i] = tmp[(size_t)r * D_EMB + i] + bo[i];
    __syncthreads();
    float s1 = 0.f, s2 = 0.f;
    for (int i = t; i < D_EMB; i += 256) { float x = row[i]; s1 += x; s2 += x * x; }
    float tot  = block_reduce_sum256(s1, red);
    float tot2 = block_reduce_sum256(s2, red);
    float mean = tot * (1.0f / D_EMB);
    float inv  = rsqrtf(tot2 * (1.0f / D_EMB) - mean * mean + 1e-5f);
    for (int i = t; i < D_EMB; i += 256)
        out[(size_t)r * D_EMB + i] = (row[i] - mean) * inv * gam[i] + bet[i];
}

// ---------------- host: numpy RandomState(42).randint(2, 2**31, 4) x2 --------
static void numpy_mults_mod_table(unsigned out[8]) {
    unsigned mt[624];
    mt[0] = 42u;
    for (unsigned i = 1; i < 624; ++i)
        mt[i] = 1812433253u * (mt[i - 1] ^ (mt[i - 1] >> 30)) + i;
    int pos = 624;
    auto next32 = [&]() -> unsigned {
        if (pos >= 624) {
            for (int i = 0; i < 624; ++i) {
                unsigned y = (mt[i] & 0x80000000u) | (mt[(i + 1) % 624] & 0x7fffffffu);
                unsigned v = mt[(i + 397) % 624] ^ (y >> 1);
                if (y & 1u) v ^= 2567483615u;
                mt[i] = v;
            }
            pos = 0;
        }
        unsigned y = mt[pos++];
        y ^= y >> 11;
        y ^= (y << 7)  & 2636928640u;
        y ^= (y << 15) & 4022730752u;
        y ^= y >> 18;
        return y;
    };
    const unsigned long long rng  = 0x7ffffffdull;  // high-low-1
    const unsigned long long mask = 0x7fffffffull;
    for (int i = 0; i < 8; ++i) {
        unsigned long long v;
        do {
            unsigned long long hi = next32(), lo = next32();
            v = ((hi << 32) | lo) & mask;
        } while (v > rng);
        out[i] = (unsigned)((2ull + v) % (unsigned long long)TABLE_SZ);
    }
}

// ---------------- launcher ----------------
extern "C" void kernel_launch(void* const* d_in, const int* in_sizes, int n_in,
                              void* d_out, int out_size, void* d_ws, size_t ws_size,
                              hipStream_t stream) {
    (void)n_in; (void)out_size; (void)ws_size;
    const int BS = in_sizes[0];            // B*S = 8192
    const int S  = SEQ_LEN;

    const int*   tok    = (const int*)d_in[0];
    const float* hidden = (const float*)d_in[1];
    const float* tables = (const float*)d_in[2];
    const float* qn_g   = (const float*)d_in[3];
    const float* qn_b   = (const float*)d_in[4];
    const float* mn_g   = (const float*)d_in[5];
    const float* mn_b   = (const float*)d_in[6];
    const float* Wq     = (const float*)d_in[7];
    const float* Wk     = (const float*)d_in[8];
    const float* Wv     = (const float*)d_in[9];
    const float* Wo     = (const float*)d_in[10];
    const float* bo     = (const float*)d_in[11];
    const float* on_g   = (const float*)d_in[12];
    const float* on_b   = (const float*)d_in[13];
    const float* conv_w = (const float*)d_in[14];
    const float* conv_b = (const float*)d_in[15];
    float* out = (float*)d_out;

    // workspace carve-up (256B aligned)
    size_t off = 0;
    auto take = [&](size_t bytes) -> void* {
        void* p = (char*)d_ws + off;
        off += (bytes + 255) & ~(size_t)255;
        return p;
    };
    const size_t BSz = (size_t)BS;
    unsigned short* ahi = (unsigned short*)take(BSz * CONCAT_DIM * 2);
    unsigned short* alo = (unsigned short*)take(BSz * CONCAT_DIM * 2);
    unsigned short* hhi = (unsigned short*)take(BSz * HID_DIM * 2);
    unsigned short* hlo = (unsigned short*)take(BSz * HID_DIM * 2);
    unsigned short* wqh = (unsigned short*)take((size_t)HID_DIM * D_EMB * 2);
    unsigned short* wql = (unsigned short*)take((size_t)HID_DIM * D_EMB * 2);
    unsigned short* wkh = (unsigned short*)take((size_t)CONCAT_DIM * D_EMB * 2);
    unsigned short* wkl = (unsigned short*)take((size_t)CONCAT_DIM * D_EMB * 2);
    unsigned short* wvh = (unsigned short*)take((size_t)CONCAT_DIM * D_EMB * 2);
    unsigned short* wvl = (unsigned short*)take((size_t)CONCAT_DIM * D_EMB * 2);
    unsigned short* woh = (unsigned short*)take((size_t)D_EMB * D_EMB * 2);
    unsigned short* wol = (unsigned short*)take((size_t)D_EMB * D_EMB * 2);
    float* qf   = (float*)take(BSz * D_EMB * 4);   // reused as tmp for Wo GEMM
    float* kf   = (float*)take(BSz * D_EMB * 4);
    float* vf   = (float*)take(BSz * D_EMB * 4);
    float* memf = (float*)take(BSz * D_EMB * 4);
    unsigned short* yhi = (unsigned short*)take(BSz * D_EMB * 2);
    unsigned short* ylo = (unsigned short*)take(BSz * D_EMB * 2);

    Mults mu;
    numpy_mults_mod_table(mu.m);

    // 1) split weights to bf16 hi/lo, transposed to [N,K]
    split_wT_kernel<<<(HID_DIM * D_EMB + 255) / 256, 256, 0, stream>>>(Wq, wqh, wql, HID_DIM, D_EMB);
    split_wT_kernel<<<(CONCAT_DIM * D_EMB + 255) / 256, 256, 0, stream>>>(Wk, wkh, wkl, CONCAT_DIM, D_EMB);
    split_wT_kernel<<<(CONCAT_DIM * D_EMB + 255) / 256, 256, 0, stream>>>(Wv, wvh, wvl, CONCAT_DIM, D_EMB);
    split_wT_kernel<<<(D_EMB * D_EMB + 255) / 256, 256, 0, stream>>>(Wo, woh, wol, D_EMB, D_EMB);

    // 2) hash + gather + LN(raw) -> bf16 hi/lo
    retrieve_ln_split_kernel<<<BS, 256, 0, stream>>>(tok, tables, mn_g, mn_b, ahi, alo, mu, S);

    // 3) LN(hidden) -> bf16 hi/lo
    ln_split_kernel<HID_DIM><<<BS, 256, 0, stream>>>(hidden, qn_g, qn_b, hhi, hlo);

    // 4-6) q, k, v GEMMs (bf16x3 WMMA, async-staged)
    dim3 gg(BS / 128, D_EMB / 128);
    gemm_bf16x3_kernel<<<gg, 256, 0, stream>>>(hhi, hlo, wqh, wql, qf, BS, D_EMB, HID_DIM);
    gemm_bf16x3_kernel<<<gg, 256, 0, stream>>>(ahi, alo, wkh, wkl, kf, BS, D_EMB, CONCAT_DIM);
    gemm_bf16x3_kernel<<<gg, 256, 0, stream>>>(ahi, alo, wvh, wvl, vf, BS, D_EMB, CONCAT_DIM);

    // 7) gate + mem  (gate written to tail of d_out)
    gate_mem_kernel<<<BS, 256, 0, stream>>>(qf, kf, vf, memf, out + BSz * D_EMB);

    // 8) causal depthwise conv + residual, split to bf16
    conv_split_kernel<<<(BS * D_EMB) / 256, 256, 0, stream>>>(memf, conv_w, conv_b, yhi, ylo, S);

    // 9) Wo GEMM (tmp aliases qf)
    gemm_bf16x3_kernel<<<gg, 256, 0, stream>>>(yhi, ylo, woh, wol, qf, BS, D_EMB, D_EMB);

    // 10) out = LN(tmp + bo)
    bias_ln_kernel<<<BS, 256, 0, stream>>>(qf, bo, on_g, on_b, out);
}
